// Encoder_51582557225690
// MI455X (gfx1250) — compile-verified, matrix operand order checked by reading, not run on hardware
//
#include <hip/hip_runtime.h>
#include <hip/hip_bf16.h>

// ---------------- model constants (match reference) ----------------
#define LAYERS 8
#define DMODEL 1024
#define NHEAD  16
#define HDIM   64
#define FFN    2048
#define SEQ    1024
#define MEMN   128
#define TTOT   1280      // SEQ + 2*MEMN
#define KVROWS 1408      // MEMN + SEQ + MEMN + MEMN

typedef __attribute__((ext_vector_type(16))) __bf16 v16bf;
typedef __attribute__((ext_vector_type(8)))  float  v8f;
typedef __attribute__((ext_vector_type(4)))  unsigned int v4u;
typedef __attribute__((ext_vector_type(8)))  int v8i;
typedef __attribute__((ext_vector_type(4)))  int v4i;

#if defined(__has_builtin)
#  if __has_builtin(__builtin_amdgcn_tensor_load_to_lds)
#    define HAVE_TDM 1
#  else
#    define HAVE_TDM 0
#  endif
#else
#  define HAVE_TDM 0
#endif

#if HAVE_TDM
// Tensor Data Mover: DMA a 64-row x 32-col fp32 tile (row stride K elements)
// from global into LDS at byte offset ldsOff. D# packed per ISA 8.3/8.4:
//  group0: count=1 | lds_addr | global_addr[31:0] | global_addr[56:32], type=2
//  group1: data_size=4B | tensor_dim0=32 | tensor_dim1=64 | tile_dim0=32 |
//          tile_dim1=64 | tensor_dim0_stride=K
__device__ __forceinline__ void tdm_load_a_tile(const float* gsrc, unsigned ldsOff,
                                                int K) {
  const unsigned long long ga = (unsigned long long)(uintptr_t)gsrc;
  v4u g0 = { 1u, ldsOff, (unsigned)ga,
             (unsigned)((ga >> 32) & 0x01ffffffu) | 0x80000000u };
  v8i g1 = { 0x20000,            // data_size = 2 (4 bytes)
             (int)(32u << 16),   // tensor_dim0[15:0]=32 in bits 63:48
             (int)(64u << 16),   // tensor_dim0 hi=0 | tensor_dim1[15:0]=64
             (int)(32u << 16),   // tensor_dim1 hi=0 | tile_dim0=32
             64,                 // tile_dim1=64 | tile_dim2=0
             K,                  // tensor_dim0_stride[31:0] (elements)
             0, 0 };
  v4i gz = { 0, 0, 0, 0 };
#if __clang_major__ >= 23
  v8i gz8 = { 0, 0, 0, 0, 0, 0, 0, 0 };
  __builtin_amdgcn_tensor_load_to_lds(g0, g1, gz, gz, gz8, 0);
#else
  __builtin_amdgcn_tensor_load_to_lds(g0, g1, gz, gz, 0);
#endif
}
#endif

// -------------------------------------------------------------------
// Fragment loaders for V_WMMA_F32_16X16X32_BF16 (wave32).
// 16-bit A-matrix 16x32 layout: lane&15 = row, (lane>>4)*8 = K base;
// element e maps to K = base + e + (e>=8 ? 8 : 0).
// Works for global or LDS source (generic pointer).
// -------------------------------------------------------------------
__device__ __forceinline__ v16bf frag_rows(const float* base, int stride, float scale) {
  const int lane = threadIdx.x & 31;
  const float* p = base + (size_t)(lane & 15) * stride + ((lane >> 4) << 3);
  float4 a0 = *(const float4*)(p + 0);
  float4 a1 = *(const float4*)(p + 4);
  float4 a2 = *(const float4*)(p + 16);
  float4 a3 = *(const float4*)(p + 20);
  v16bf r;
  r[0] = (__bf16)(a0.x * scale); r[1] = (__bf16)(a0.y * scale);
  r[2] = (__bf16)(a0.z * scale); r[3] = (__bf16)(a0.w * scale);
  r[4] = (__bf16)(a1.x * scale); r[5] = (__bf16)(a1.y * scale);
  r[6] = (__bf16)(a1.z * scale); r[7] = (__bf16)(a1.w * scale);
  r[8] = (__bf16)(a2.x * scale); r[9] = (__bf16)(a2.y * scale);
  r[10] = (__bf16)(a2.z * scale); r[11] = (__bf16)(a2.w * scale);
  r[12] = (__bf16)(a3.x * scale); r[13] = (__bf16)(a3.y * scale);
  r[14] = (__bf16)(a3.z * scale); r[15] = (__bf16)(a3.w * scale);
  return r;
}

// B fragment when B is stored K-major in global (weights: [K x N] row-major).
// lane&15 = column, rows strided by ldw.
__device__ __forceinline__ v16bf frag_cols(const float* __restrict__ W, int ldw) {
  const int lane = threadIdx.x & 31;
  const float* p = W + (size_t)((lane >> 4) << 3) * ldw + (lane & 15);
  v16bf r;
#pragma unroll
  for (int e = 0; e < 8; ++e) r[e] = (__bf16)p[(size_t)e * ldw];
#pragma unroll
  for (int e = 0; e < 8; ++e) r[8 + e] = (__bf16)p[(size_t)(16 + e) * ldw];
  return r;
}

// -------------------------------------------------------------------
// GEMM: C[T,N] (+)= A[T,K] * W[K,N], fp32 in memory, bf16 WMMA compute.
// Block = 128 threads = 4 waves; C tile 64x64; wave tile 16x64.
//  - A tile (64x32 fp32) DMA'd by the Tensor Data Mover (TENSORcnt)
//  - W tile (32x64 fp32) staged by async-LDS loads (ASYNCcnt)
// Both double-buffered so tile k+1 streams behind the WMMAs of tile k.
// -------------------------------------------------------------------
template <bool ADD>
__global__ void k_gemm_bf16(const float* __restrict__ A, const float* __restrict__ W,
                            float* __restrict__ C, int K, int N) {
  __shared__ __align__(16) float ash[2][64][32];   // 2 x 8KB A tiles
  __shared__ __align__(16) float bsh[2][32][64];   // 2 x 8KB B tiles
  const int lane = threadIdx.x & 31;
  const int wave = threadIdx.x >> 5;
  const int rowBlk = blockIdx.y * 64;
  const int colBase = blockIdx.x * 64;

  // async-stage one 32x64 fp32 tile of W (512 chunks of 16B, 4 per thread)
  auto stageB = [&](int k0, int buf) {
#pragma unroll
    for (int i = 0; i < 4; ++i) {
      const int c   = (int)threadIdx.x + 128 * i;
      const int r   = c >> 4;
      const int col = (c & 15) << 2;
      const float* g = W + (size_t)(k0 + r) * N + colBase + col;
      const unsigned l = (unsigned)(uintptr_t)(&bsh[buf][r][col]);
      asm volatile("global_load_async_to_lds_b128 %0, %1, off"
                   :: "v"(l), "v"(g) : "memory");
    }
  };
  // stage one 64x32 fp32 tile of A
  auto stageA = [&](int k0, int buf) {
#if HAVE_TDM
    if (wave == 0)
      tdm_load_a_tile(A + (size_t)rowBlk * K + k0,
                      (unsigned)(uintptr_t)(&ash[buf][0][0]), K);
#else
#pragma unroll
    for (int i = 0; i < 4; ++i) {
      const int c   = (int)threadIdx.x + 128 * i;
      const int r   = c >> 3;
      const int col = (c & 7) << 2;
      const float* g = A + (size_t)(rowBlk + r) * K + k0 + col;
      const unsigned l = (unsigned)(uintptr_t)(&ash[buf][r][col]);
      asm volatile("global_load_async_to_lds_b128 %0, %1, off"
                   :: "v"(l), "v"(g) : "memory");
    }
#endif
  };

  stageB(0, 0);
  stageA(0, 0);
  v8f acc[4] = {};
  for (int k0 = 0; k0 < K; k0 += 32) {
    const int buf = (k0 >> 5) & 1;
    if (k0 + 32 < K) {
      stageB(k0 + 32, buf ^ 1);
      stageA(k0 + 32, buf ^ 1);
      // async/tensor ops complete in order: allowing the prefetch to remain
      // outstanding drains exactly the current buffer's transfers.
#if HAVE_TDM
      asm volatile("s_wait_asynccnt 4" ::: "memory");
      if (wave == 0) __builtin_amdgcn_s_wait_tensorcnt(1);
#else
      asm volatile("s_wait_asynccnt 8" ::: "memory");
#endif
    } else {
      asm volatile("s_wait_asynccnt 0" ::: "memory");
#if HAVE_TDM
      if (wave == 0) __builtin_amdgcn_s_wait_tensorcnt(0);
#endif
    }
    __syncthreads();   // staged tiles visible to all waves

    v16bf a = frag_rows(&ash[buf][wave * 16][0], 32, 1.0f);
#pragma unroll
    for (int ns = 0; ns < 4; ++ns) {
      // build B fragment (32x16) from the raw fp32 LDS tile
      const int n  = ns * 16 + (lane & 15);
      const int kb = (lane >> 4) << 3;
      v16bf b;
#pragma unroll
      for (int e = 0; e < 8; ++e) b[e]     = (__bf16)bsh[buf][kb + e][n];
#pragma unroll
      for (int e = 0; e < 8; ++e) b[8 + e] = (__bf16)bsh[buf][kb + 16 + e][n];
      acc[ns] = __builtin_amdgcn_wmma_f32_16x16x32_bf16(
          false, a, false, b, (short)0, acc[ns], false, false);
    }
    __syncthreads();   // all waves done reading before buffers are reused
  }

  // D layout: lane&15 = col, rows (lane>>4)*8 + r
  const int cn = colBase + (lane & 15);
  const int rm = rowBlk + wave * 16 + ((lane >> 4) << 3);
#pragma unroll
  for (int ns = 0; ns < 4; ++ns) {
#pragma unroll
    for (int r = 0; r < 8; ++r) {
      float* cp = C + (size_t)(rm + r) * N + cn + ns * 16;
      if (ADD) *cp += acc[ns][r]; else *cp = acc[ns][r];
    }
  }
}

// -------------------------------------------------------------------
// Flash attention with WMMA. One wave = one 16-row Q tile over all keys.
// Keys live in [Kmain,Vmain) for LkMain rows, then [Ktail,Vtail) for
// LkTail rows (used by the forget branch whose tail is non-contiguous).
// Mask: key j visible to query q iff j <= q + causalStart.
// -------------------------------------------------------------------
__global__ void k_attn(const float* __restrict__ Q,
                       const float* __restrict__ Kmain, const float* __restrict__ Vmain,
                       int LkMain,
                       const float* __restrict__ Ktail, const float* __restrict__ Vtail,
                       int LkTail,
                       float* __restrict__ O, int causalStart) {
  __shared__ float                ssc[4][16][32];
  __shared__ __align__(16) __bf16 pls[4][16][32];

  const int lane = threadIdx.x & 31;
  const int wave = threadIdx.x >> 5;
  const int hh   = lane >> 4;       // half-wave id
  const int m    = lane & 15;       // my softmax row
  const int h64  = blockIdx.y * HDIM;
  const int qBase = (blockIdx.x * 4 + wave) * 16;
  const int qIdx  = qBase + m;

  const float* qp = Q + (size_t)qBase * DMODEL + h64;
  v16bf q0 = frag_rows(qp, DMODEL, 0.125f);        // 1/sqrt(HDIM) folded in
  v16bf q1 = frag_rows(qp + 32, DMODEL, 0.125f);

  v8f acc[4] = {};
  float mrow = -1e30f, lrow = 0.f;
  const int Lk = LkMain + LkTail;

  for (int j0 = 0; j0 < Lk; j0 += 32) {
    if (j0 > qBase + 15 + causalStart) break;       // uniform per wave
    const float* kr; const float* vr;
    if (j0 < LkMain) { kr = Kmain + (size_t)j0 * DMODEL + h64;
                       vr = Vmain + (size_t)j0 * DMODEL + h64; }
    else             { kr = Ktail + (size_t)(j0 - LkMain) * DMODEL + h64;
                       vr = Vtail + (size_t)(j0 - LkMain) * DMODEL + h64; }

    // scores: two 16x16 tiles, contraction HDIM=64 as 2x32
#pragma unroll
    for (int t = 0; t < 2; ++t) {
      v16bf kb0 = frag_rows(kr + (size_t)t * 16 * DMODEL, DMODEL, 1.0f);
      v16bf kb1 = frag_rows(kr + (size_t)t * 16 * DMODEL + 32, DMODEL, 1.0f);
      v8f s = {};
      s = __builtin_amdgcn_wmma_f32_16x16x32_bf16(false, q0, false, kb0, (short)0, s, false, false);
      s = __builtin_amdgcn_wmma_f32_16x16x32_bf16(false, q1, false, kb1, (short)0, s, false, false);
#pragma unroll
      for (int r = 0; r < 8; ++r)
        ssc[wave][hh * 8 + r][t * 16 + m] = s[r];
    }

    // online softmax: lane handles row m, columns hh*16..hh*16+15
    float sc[16];
    float cm = -1e30f;
#pragma unroll
    for (int c = 0; c < 16; ++c) {
      float v = ssc[wave][m][hh * 16 + c];
      if (j0 + hh * 16 + c > qIdx + causalStart) v = -1e30f;
      sc[c] = v;
      cm = fmaxf(cm, v);
    }
    cm = fmaxf(cm, __shfl_xor(cm, 16, 32));
    const float mnew = fmaxf(mrow, cm);
    const float corr = __expf(mrow - mnew);
    float ps = 0.f;
#pragma unroll
    for (int c = 0; c < 16; ++c) {
      float p = __expf(sc[c] - mnew);
      ps += p;
      pls[wave][m][hh * 16 + c] = (__bf16)p;
    }
    ps += __shfl_xor(ps, 16, 32);
    lrow = lrow * corr + ps;
    mrow = mnew;

    // rescale running accumulators (gather per-row corr via shuffle)
#pragma unroll
    for (int r = 0; r < 8; ++r) {
      const float cr = __shfl(corr, hh * 8 + r, 32);
#pragma unroll
      for (int f = 0; f < 4; ++f) acc[f][r] *= cr;
    }

    // P fragment (A layout) back from LDS: 2x 16B vector reads per lane
    v16bf pf;
    {
      const __bf16* pb = &pls[wave][m][hh * 8];
      ((uint4*)&pf)[0] = *(const uint4*)(pb);
      ((uint4*)&pf)[1] = *(const uint4*)(pb + 16);
    }
    // P(16x32) x V(32x64)
#pragma unroll
    for (int f = 0; f < 4; ++f) {
      v16bf vb = frag_cols(vr + f * 16, DMODEL);
      acc[f] = __builtin_amdgcn_wmma_f32_16x16x32_bf16(
          false, pf, false, vb, (short)0, acc[f], false, false);
    }
  }

  const float inv = 1.f / lrow;
#pragma unroll
  for (int r = 0; r < 8; ++r) {
    const float iv = __shfl(inv, hh * 8 + r, 32);
    float* op = O + (size_t)(qBase + hh * 8 + r) * DMODEL + h64 + (lane & 15);
#pragma unroll
    for (int f = 0; f < 4; ++f) op[f * 16] = acc[f][r] * iv;
  }
}

// -------------------------------------------------------------------
// RMSNorm: one row (D=1024) per 256-thread block.
// -------------------------------------------------------------------
__global__ void k_rmsnorm(const float* __restrict__ X, const float* __restrict__ w,
                          float* __restrict__ Y) {
  __shared__ float red[8];
  const int row = blockIdx.x;
  const float* x = X + (size_t)row * DMODEL;
  float ss = 0.f;
  for (int i = threadIdx.x; i < DMODEL; i += 256) { float v = x[i]; ss += v * v; }
#pragma unroll
  for (int o = 16; o > 0; o >>= 1) ss += __shfl_xor(ss, o, 32);
  if ((threadIdx.x & 31) == 0) red[threadIdx.x >> 5] = ss;
  __syncthreads();
  if (threadIdx.x < 8) {
    float v = red[threadIdx.x];
#pragma unroll
    for (int o = 4; o > 0; o >>= 1) v += __shfl_xor(v, o, 32);
    if (threadIdx.x == 0) red[0] = v;
  }
  __syncthreads();
  const float rs = rsqrtf(red[0] * (1.f / DMODEL) + 1e-5f);
  float* y = Y + (size_t)row * DMODEL;
  for (int i = threadIdx.x; i < DMODEL; i += 256) y[i] = x[i] * w[i] * rs;
}

// -------------------------------------------------------------------
// RoPE in place: one row per block, 512 threads (16 heads x 32 pairs).
// -------------------------------------------------------------------
__global__ void k_rope(float* __restrict__ X, int pos0) {
  const int row = blockIdx.x;
  float* x = X + (size_t)row * DMODEL;
  const int h = threadIdx.x >> 5;
  const int i = threadIdx.x & 31;
  const float ang = (float)(pos0 + row) * __powf(10000.f, -(float)(2 * i) * (1.f / HDIM));
  float s, c;
  __sincosf(ang, &s, &c);
  const float x0 = x[h * HDIM + i];
  const float x1 = x[h * HDIM + i + 32];
  x[h * HDIM + i]      = x0 * c - x1 * s;
  x[h * HDIM + i + 32] = x1 * c + x0 * s;
}

// -------------------------------------------------------------------
// Input assembly / small elementwise kernels.
// -------------------------------------------------------------------
__global__ void k_build_input(const int* __restrict__ ids, const float* __restrict__ embed,
                              const float* __restrict__ beacon, const float* __restrict__ forget,
                              float* __restrict__ Xcat) {
  const int t = blockIdx.x;
  const float* src;
  if (t < SEQ)             src = embed + (size_t)ids[t] * DMODEL;
  else if (t < SEQ + MEMN) src = beacon + (size_t)(t - SEQ) * DMODEL;
  else                     src = forget + (size_t)(t - SEQ - MEMN) * DMODEL;
  float* dst = Xcat + (size_t)t * DMODEL;
  for (int i = threadIdx.x; i < DMODEL; i += 256) dst[i] = src[i];
}

__global__ void k_copy(const float* __restrict__ src, float* __restrict__ dst, int n) {
  const int i = blockIdx.x * blockDim.x + threadIdx.x;
  if (i < n) dst[i] = src[i];
}

__global__ void k_silu_mul(float* __restrict__ G, const float* __restrict__ U, int n) {
  const int i = blockIdx.x * blockDim.x + threadIdx.x;
  if (i < n) { const float g = G[i]; G[i] = (g / (1.f + __expf(-g))) * U[i]; }
}

__global__ void k_final_gate(const float* __restrict__ mem, const float* __restrict__ inj,
                             const float* __restrict__ fg, float* __restrict__ out, int n) {
  const int i = blockIdx.x * blockDim.x + threadIdx.x;
  if (i < n) {
    const float g = 1.f / (1.f + __expf(-fg[i]));
    out[i] = mem[i] * g + inj[i] * (1.f - g);
  }
}

// -------------------------------------------------------------------
extern "C" void kernel_launch(void* const* d_in, const int* in_sizes, int n_in,
                              void* d_out, int out_size, void* d_ws, size_t ws_size,
                              hipStream_t stream) {
  (void)in_sizes; (void)n_in; (void)out_size; (void)ws_size;
  const int*   ids    = (const int*)d_in[0];
  const float* memory = (const float*)d_in[1];
  const float* beacon = (const float*)d_in[2];
  const float* forget = (const float*)d_in[3];
  const float* embed  = (const float*)d_in[4];
  const float* ln1    = (const float*)d_in[5];
  const float* ln2    = (const float*)d_in[6];
  const float* Wq  = (const float*)d_in[7];
  const float* Wk  = (const float*)d_in[8];
  const float* Wv  = (const float*)d_in[9];
  const float* Wo  = (const float*)d_in[10];
  const float* mWk = (const float*)d_in[11];
  const float* mWv = (const float*)d_in[12];
  const float* bWq = (const float*)d_in[13];
  const float* bWk = (const float*)d_in[14];
  const float* bWv = (const float*)d_in[15];
  const float* fWq = (const float*)d_in[16];
  const float* fWk = (const float*)d_in[17];
  const float* fWv = (const float*)d_in[18];
  const float* Wg  = (const float*)d_in[19];
  const float* Wu  = (const float*)d_in[20];
  const float* Wd  = (const float*)d_in[21];

  float* ws   = (float*)d_ws;
  float* Xcat = ws; ws += (size_t)TTOT * DMODEL;     // residual stream (cat)
  float* Xn   = ws; ws += (size_t)TTOT * DMODEL;     // normalized activations
  float* Qb   = ws; ws += (size_t)TTOT * DMODEL;     // queries (qh | bq | fq)
  float* Kb   = ws; ws += (size_t)KVROWS * DMODEL;   // keys (mk | kh | bk | fk)
  float* Vb   = ws; ws += (size_t)KVROWS * DMODEL;   // values (mv | vh | bv | fv)
  float* Ob   = ws; ws += (size_t)TTOT * DMODEL;     // attention output pre-Wo
  float* Gb   = ws; ws += (size_t)TTOT * FFN;        // gate / fused hidden
  float* Ub   = ws; ws += (size_t)TTOT * FFN;        // up
  float* injb = ws; ws += (size_t)LAYERS * MEMN * DMODEL;
  float* fgb  = ws; ws += (size_t)LAYERS * MEMN * DMODEL;

  auto gemm = [&](const float* A, const float* W, float* C, int Trows, int K, int N,
                  bool add) {
    dim3 g(N / 64, Trows / 64);
    if (add) k_gemm_bf16<true><<<g, 128, 0, stream>>>(A, W, C, K, N);
    else     k_gemm_bf16<false><<<g, 128, 0, stream>>>(A, W, C, K, N);
  };

  const size_t dd = (size_t)DMODEL * DMODEL;
  const size_t df = (size_t)DMODEL * FFN;
  const size_t md = (size_t)MEMN * DMODEL;

  k_build_input<<<TTOT, 256, 0, stream>>>(ids, embed, beacon, forget, Xcat);

  for (int l = 0; l < LAYERS; ++l) {
    const float* meml = memory + (size_t)l * md;

    // record beacon/forget state entering this layer (scan outputs)
    k_copy<<<(MEMN * DMODEL + 255) / 256, 256, 0, stream>>>(
        Xcat + (size_t)SEQ * DMODEL, injb + (size_t)l * md, MEMN * DMODEL);
    k_copy<<<(MEMN * DMODEL + 255) / 256, 256, 0, stream>>>(
        Xcat + (size_t)(SEQ + MEMN) * DMODEL, fgb + (size_t)l * md, MEMN * DMODEL);

    k_rmsnorm<<<TTOT, 256, 0, stream>>>(Xcat, ln1 + (size_t)l * DMODEL, Xn);

    // ---- projections (bf16 WMMA GEMMs) ----
    gemm(Xn, Wq + l * dd, Qb, SEQ, DMODEL, DMODEL, false);
    gemm(Xn + (size_t)SEQ * DMODEL, bWq + l * dd, Qb + (size_t)SEQ * DMODEL,
         MEMN, DMODEL, DMODEL, false);
    gemm(Xn + (size_t)(SEQ + MEMN) * DMODEL, fWq + l * dd,
         Qb + (size_t)(SEQ + MEMN) * DMODEL, MEMN, DMODEL, DMODEL, false);

    gemm(meml, mWk + l * dd, Kb, MEMN, DMODEL, DMODEL, false);
    gemm(Xn, Wk + l * dd, Kb + md, SEQ, DMODEL, DMODEL, false);
    gemm(Xn + (size_t)SEQ * DMODEL, bWk + l * dd,
         Kb + (size_t)(MEMN + SEQ) * DMODEL, MEMN, DMODEL, DMODEL, false);
    gemm(Xn + (size_t)(SEQ + MEMN) * DMODEL, fWk + l * dd,
         Kb + (size_t)(MEMN + SEQ + MEMN) * DMODEL, MEMN, DMODEL, DMODEL, false);

    gemm(meml, mWv + l * dd, Vb, MEMN, DMODEL, DMODEL, false);
    gemm(Xn, Wv + l * dd, Vb + md, SEQ, DMODEL, DMODEL, false);
    gemm(Xn + (size_t)SEQ * DMODEL, bWv + l * dd,
         Vb + (size_t)(MEMN + SEQ) * DMODEL, MEMN, DMODEL, DMODEL, false);
    gemm(Xn + (size_t)(SEQ + MEMN) * DMODEL, fWv + l * dd,
         Vb + (size_t)(MEMN + SEQ + MEMN) * DMODEL, MEMN, DMODEL, DMODEL, false);

    // ---- RoPE ----
    k_rope<<<SEQ, 512, 0, stream>>>(Qb, MEMN);                                        // qh
    k_rope<<<MEMN, 512, 0, stream>>>(Qb + (size_t)SEQ * DMODEL, MEMN + SEQ);          // bq
    k_rope<<<MEMN, 512, 0, stream>>>(Qb + (size_t)(SEQ + MEMN) * DMODEL, MEMN + SEQ); // fq
    k_rope<<<MEMN, 512, 0, stream>>>(Kb, 0);                                          // mk
    k_rope<<<SEQ, 512, 0, stream>>>(Kb + md, MEMN);                                   // kh
    k_rope<<<MEMN, 512, 0, stream>>>(Kb + (size_t)(MEMN + SEQ) * DMODEL, MEMN + SEQ); // bk
    k_rope<<<MEMN, 512, 0, stream>>>(Kb + (size_t)(MEMN + SEQ + MEMN) * DMODEL,
                                     MEMN + SEQ);                                     // fk

    // ---- attention (WMMA flash) ----
    {
      dim3 gh(SEQ / 64, NHEAD);
      k_attn<<<gh, 128, 0, stream>>>(Qb, Kb, Vb, MEMN + SEQ,
                                     (const float*)nullptr, (const float*)nullptr, 0,
                                     Ob, MEMN);
      dim3 gg(MEMN / 64, NHEAD);
      k_attn<<<gg, 128, 0, stream>>>(Qb + (size_t)SEQ * DMODEL, Kb, Vb,
                                     MEMN + SEQ + MEMN,
                                     (const float*)nullptr, (const float*)nullptr, 0,
                                     Ob + (size_t)SEQ * DMODEL, MEMN + SEQ);
      k_attn<<<gg, 128, 0, stream>>>(Qb + (size_t)(SEQ + MEMN) * DMODEL, Kb, Vb,
                                     MEMN + SEQ,
                                     Kb + (size_t)(MEMN + SEQ + MEMN) * DMODEL,
                                     Vb + (size_t)(MEMN + SEQ + MEMN) * DMODEL, MEMN,
                                     Ob + (size_t)(SEQ + MEMN) * DMODEL, MEMN + SEQ);
    }

    // ---- output projection + residual ----
    gemm(Ob, Wo + l * dd, Xcat, TTOT, DMODEL, DMODEL, true);

    // ---- MLP (SwiGLU) ----
    k_rmsnorm<<<TTOT, 256, 0, stream>>>(Xcat, ln2 + (size_t)l * DMODEL, Xn);
    gemm(Xn, Wg + l * df, Gb, TTOT, DMODEL, FFN, false);
    gemm(Xn, Wu + l * df, Ub, TTOT, DMODEL, FFN, false);
    k_silu_mul<<<(TTOT * FFN + 255) / 256, 256, 0, stream>>>(Gb, Ub, TTOT * FFN);
    gemm(Gb, Wd + l * df, Xcat, TTOT, FFN, DMODEL, true);
  }

  const int n = LAYERS * MEMN * DMODEL;
  k_final_gate<<<(n + 255) / 256, 256, 0, stream>>>(memory, injb, fgb,
                                                    (float*)d_out, n);
}
